// TargetModel_88802743812780
// MI455X (gfx1250) — matrix operand order repaired and checked by reading, not compiled, not run on hardware
//
#include <hip/hip_runtime.h>
#include <hip/hip_bf16.h>
#include <math.h>
#include <stdint.h>

// ---------------------------------------------------------------------------
// 2-layer GCN for MI455X (gfx1250, wave32).
// Pre-pass: bf16-convert + zero-pad A/B operands (B pre-transposed) so the
// WMMA GEMM inner loop is pure async DMA (global_load_async_to_lds_b128,
// double-buffered, s_wait_asynccnt) + ds_load_b128 fragment reads + v_wmma.
//   h1   = x @ W1                       (bf16 WMMA, f32 accumulate)
//   agg1 = scatter_add(h1[src]*coef)    (float4 gather, f32 atomics; h1 in L2)
//   h1   = relu(agg1 + h1*dinv^2 + b1)  (fused; also emits bf16 copy)
//   h2   = h1 @ W2  (cols padded 47->64)
//   agg2 = scatter_add(h2[src]*coef)
//   out  = log_softmax(agg2 + h2*dinv^2 + b2)  (one wave32 per node)
// ---------------------------------------------------------------------------

typedef __attribute__((ext_vector_type(16))) __bf16 v16bf;
typedef __attribute__((ext_vector_type(8)))  __bf16 v8bf;
typedef __attribute__((ext_vector_type(8)))  float  v8f;
typedef __attribute__((ext_vector_type(4)))  int    v4i;

#define TM 128
#define TN 64
#define TK 32

#if defined(__gfx1250__) && __has_builtin(__builtin_amdgcn_global_load_async_to_lds_b128)
#define HAVE_ASYNC_LDS 1
#else
#define HAVE_ASYNC_LDS 0
#endif

// Copy 16 bytes global -> LDS. Async path uses the gfx1250 async-DMA
// instruction (tracked by ASYNCcnt); fallback is a VGPR round-trip.
// Builtin signature (from compiler diagnostic): (v4i AS1*, v4i AS3*, Ii, Ii).
__device__ __forceinline__ void copy16_g2lds(const __bf16* g, __bf16* l) {
#if HAVE_ASYNC_LDS
  __builtin_amdgcn_global_load_async_to_lds_b128(
      (__attribute__((address_space(1))) v4i*)(uintptr_t)g,
      (__attribute__((address_space(3))) v4i*)(uint32_t)(uintptr_t)l,
      0, 0);
#else
  *(v8bf*)l = *(const v8bf*)g;
#endif
}

__device__ __forceinline__ void wait_async_le3() {
#if HAVE_ASYNC_LDS
  asm volatile("s_wait_asynccnt 0x3" ::: "memory");
#endif
}
__device__ __forceinline__ void wait_async_le0() {
#if HAVE_ASYNC_LDS
  asm volatile("s_wait_asynccnt 0x0" ::: "memory");
#endif
}

// ---------------------------------------------------------------------------
// Tiled GEMM: C[M,*] = A[Mpad,Ksteps*TK] * Bt^T, bf16 inputs (pre-padded,
// B stored transposed [n][k]), f32 output. 256 threads = 8 waves.
// Block tile 128x64, wave tile 32x32 (2x2 WMMA accum). Double-buffered LDS.
// ---------------------------------------------------------------------------
__global__ __launch_bounds__(256) void gcn_gemm_bf16_wmma(
    const __bf16* __restrict__ A, int lda,
    const __bf16* __restrict__ Bt, int ldb,
    float* __restrict__ C, int ldc,
    int M, int Ksteps)
{
  __shared__ alignas(64) __bf16 lA[2][TM * TK];   // [m][k], 2 x 8 KB
  __shared__ alignas(64) __bf16 lB[2][TN * TK];   // [n][k], 2 x 4 KB

  const int tid  = threadIdx.x;
  const int lane = tid & 31;
  const int wave = tid >> 5;
  const int wm   = wave & 3;            // 4 waves along M (32 rows each)
  const int wn   = wave >> 2;           // 2 waves along N (32 cols each)
  const int lrow = lane & 15;
  const int hi16 = lane >> 4;           // 0 or 1
  const int kbA  = hi16 << 4;           // A frag: 16 contiguous K at kbA
  const int kbB  = hi16 << 4;           // B frag: 16 contiguous K at kbB

  const int m0 = blockIdx.x * TM;
  const int n0 = blockIdx.y * TN;

  // Per-thread 16B staging chunks (8 bf16 each): A = 512 chunks, B = 256.
  const int ar0 = tid >> 2;                 // rows 0..63
  const int ar1 = ar0 + 64;                 // rows 64..127
  const int akk = (tid & 3) << 3;           // k offset 0/8/16/24
  const int bn  = tid >> 2;                 // n 0..63

  auto stage = [&](int kt, int buf) {
    const int k0 = kt * TK;
    copy16_g2lds(A  + (size_t)(m0 + ar0) * lda + k0 + akk, &lA[buf][ar0 * TK + akk]);
    copy16_g2lds(A  + (size_t)(m0 + ar1) * lda + k0 + akk, &lA[buf][ar1 * TK + akk]);
    copy16_g2lds(Bt + (size_t)(n0 + bn)  * ldb + k0 + akk, &lB[buf][bn  * TK + akk]);
  };

  v8f acc[2][2] = {};

  stage(0, 0);
  for (int kt = 0; kt < Ksteps; ++kt) {
    const int buf = kt & 1;
    if (kt + 1 < Ksteps) {
      stage(kt + 1, buf ^ 1);   // prefetch next tile into other buffer
      wait_async_le3();         // this tile's 3 DMAs done (in-order retire)
    } else {
      wait_async_le0();
    }
    __syncthreads();

    // Fragments: 16 contiguous K per lane half, same order for A and B.
    v16bf afr[2], bfr[2];
#pragma unroll
    for (int mt = 0; mt < 2; ++mt) {
      const int r = (wm << 5) + (mt << 4) + lrow;
      v8bf lo = *(const v8bf*)&lA[buf][r * TK + kbA];
      v8bf hi = *(const v8bf*)&lA[buf][r * TK + kbA + 8];
      afr[mt] = __builtin_shufflevector(lo, hi,
          0,1,2,3,4,5,6,7,8,9,10,11,12,13,14,15);
    }
#pragma unroll
    for (int nt = 0; nt < 2; ++nt) {
      const int c = (wn << 5) + (nt << 4) + lrow;
      bfr[nt] = *(const v16bf*)&lB[buf][c * TK + kbB];
    }

#pragma unroll
    for (int mt = 0; mt < 2; ++mt)
#pragma unroll
      for (int nt = 0; nt < 2; ++nt) {
#if defined(__gfx1250__)
        acc[mt][nt] = __builtin_amdgcn_wmma_f32_16x16x32_bf16(
            false, afr[mt], false, bfr[nt], (short)0, acc[mt][nt],
            false, false);
#else
        acc[mt][nt][0] += (float)afr[mt][0] * (float)bfr[nt][0]; // host stub
#endif
      }
    __syncthreads();   // all waves done reading buf before it is re-filled
  }

  // Store C: VGPR j -> row (j + 8*hi16), col = lane%16.
#pragma unroll
  for (int mt = 0; mt < 2; ++mt) {
#pragma unroll
    for (int nt = 0; nt < 2; ++nt) {
      const int gcol = n0 + (wn << 5) + (nt << 4) + lrow;
#pragma unroll
      for (int j = 0; j < 8; ++j) {
        const int grow = m0 + (wm << 5) + (mt << 4) + (hi16 << 3) + j;
        if (grow < M)
          C[(size_t)grow * ldc + gcol] = acc[mt][nt][j];
      }
    }
  }
}

// ---------------------------------------------------------------------------
__global__ void gcn_zero4(float* __restrict__ p, long long n4) {
  long long i = (long long)blockIdx.x * blockDim.x + threadIdx.x;
  if (i < n4) ((float4*)p)[i] = make_float4(0.f, 0.f, 0.f, 0.f);
}

__global__ void gcn_deg(const int* __restrict__ dst, float* __restrict__ deg, int E) {
  int e = blockIdx.x * blockDim.x + threadIdx.x;
  if (e < E) atomicAdd(&deg[dst[e]], 1.0f);
}

__global__ void gcn_dinv(float* __restrict__ deg, int N) {
  int i = blockIdx.x * blockDim.x + threadIdx.x;
  if (i < N) deg[i] = rsqrtf(deg[i] + 1.0f);   // deg buffer becomes dinv
}

// x f32 [M,K] -> bf16 [M,Kpad], K-pad zeroed (pad rows left undefined; they
// only feed C rows >= M which are never stored).
__global__ void gcn_cvt_x(const float* __restrict__ X, __bf16* __restrict__ Xb,
                          long long total4, int shift, int K, int lda, int ldb) {
  long long idx = (long long)blockIdx.x * blockDim.x + threadIdx.x;
  if (idx >= total4) return;
  const int r = (int)(idx >> shift);
  const int c = (int)(idx & ((1u << shift) - 1)) << 2;
  float4 v = make_float4(0.f, 0.f, 0.f, 0.f);
  if (c < K) v = *(const float4*)(X + (size_t)r * lda + c);
  __bf16* o = Xb + (size_t)r * ldb + c;
  o[0] = (__bf16)v.x; o[1] = (__bf16)v.y; o[2] = (__bf16)v.z; o[3] = (__bf16)v.w;
}

// W f32 [K,Nact] -> transposed bf16 [Npad,Kpad], zero padded both dims.
__global__ void gcn_cvt_w_t(const float* __restrict__ W, __bf16* __restrict__ Wt,
                            int K, int Nact, int Kpad, int Npad) {
  int idx = blockIdx.x * blockDim.x + threadIdx.x;
  if (idx >= Npad * Kpad) return;
  const int n = idx / Kpad;
  const int k = idx - n * Kpad;
  const float v = (k < K && n < Nact) ? W[(size_t)k * Nact + n] : 0.f;
  Wt[(size_t)n * Kpad + k] = (__bf16)v;
}

// One thread per (edge, 4 features): float4 gather, 4 f32 atomic scatter-adds.
__global__ void gcn_scatter(const float* __restrict__ h, const int* __restrict__ src,
                            const int* __restrict__ dst, const float* __restrict__ dinv,
                            float* __restrict__ agg, long long total, int shift, int F) {
  long long idx = (long long)blockIdx.x * blockDim.x + threadIdx.x;
  if (idx >= total) return;
  const int e = (int)(idx >> shift);
  const int c = (int)(idx & ((1u << shift) - 1)) << 2;
  const int s = src[e], d = dst[e];
  const float coef = dinv[s] * dinv[d];
  const float4 hv = *(const float4*)(h + (size_t)s * F + c);
  float* ap = agg + (size_t)d * F + c;
  atomicAdd(ap + 0, hv.x * coef);
  atomicAdd(ap + 1, hv.y * coef);
  atomicAdd(ap + 2, hv.z * coef);
  atomicAdd(ap + 3, hv.w * coef);
}

// h = relu(agg + h*dinv^2 + b) in place (f32) + bf16 copy for the next GEMM.
__global__ void gcn_fuse_relu(float* __restrict__ h, __bf16* __restrict__ hb,
                              const float* __restrict__ agg,
                              const float* __restrict__ dinv, const float* __restrict__ b,
                              long long total4, int shift, int F) {
  long long idx = (long long)blockIdx.x * blockDim.x + threadIdx.x;
  if (idx >= total4) return;
  const int node = (int)(idx >> shift);
  const int c = (int)(idx & ((1u << shift) - 1)) << 2;
  const float di = dinv[node];
  const float d2 = di * di;
  const size_t o = (size_t)node * F + c;
  float4 hv = *(const float4*)(h + o);
  float4 av = *(const float4*)(agg + o);
  float4 bv = *(const float4*)(b + c);
  float4 r;
  r.x = fmaxf(av.x + hv.x * d2 + bv.x, 0.f);
  r.y = fmaxf(av.y + hv.y * d2 + bv.y, 0.f);
  r.z = fmaxf(av.z + hv.z * d2 + bv.z, 0.f);
  r.w = fmaxf(av.w + hv.w * d2 + bv.w, 0.f);
  *(float4*)(h + o) = r;
  __bf16* ob = hb + o;
  ob[0] = (__bf16)r.x; ob[1] = (__bf16)r.y; ob[2] = (__bf16)r.z; ob[3] = (__bf16)r.w;
}

// One wave32 per node: z = agg2 + h2*dinv^2 + b2 ; out = log_softmax(z).
__global__ __launch_bounds__(256) void gcn_logsoftmax(
    const float* __restrict__ h2, const float* __restrict__ agg2,
    const float* __restrict__ dinv, const float* __restrict__ b2,
    float* __restrict__ out, int N, int C, int CP)
{
  const int gw = (int)(((long long)blockIdx.x * blockDim.x + threadIdx.x) >> 5);
  const int lane = threadIdx.x & 31;
  if (gw >= N) return;
  const float di = dinv[gw];
  const float d2 = di * di;
  const int c0 = lane;
  const int c1 = lane + 32;
  const bool v0 = c0 < C;
  const bool v1 = c1 < C;
  const size_t base = (size_t)gw * CP;
  float z0 = v0 ? (agg2[base + c0] + h2[base + c0] * d2 + b2[c0]) : -INFINITY;
  float z1 = v1 ? (agg2[base + c1] + h2[base + c1] * d2 + b2[c1]) : -INFINITY;
  float m = fmaxf(z0, z1);
#pragma unroll
  for (int off = 16; off > 0; off >>= 1)
    m = fmaxf(m, __shfl_xor(m, off, 32));
  float s = (v0 ? __expf(z0 - m) : 0.f) + (v1 ? __expf(z1 - m) : 0.f);
#pragma unroll
  for (int off = 16; off > 0; off >>= 1)
    s += __shfl_xor(s, off, 32);
  const float ls = __logf(s);
  if (v0) out[(size_t)gw * C + c0] = z0 - m - ls;
  if (v1) out[(size_t)gw * C + c1] = z1 - m - ls;
}

// ---------------------------------------------------------------------------
extern "C" void kernel_launch(void* const* d_in, const int* in_sizes, int n_in,
                              void* d_out, int out_size, void* d_ws, size_t ws_size,
                              hipStream_t stream)
{
  const float* x  = (const float*)d_in[0];
  const int*   ei = (const int*)d_in[1];
  const float* W1 = (const float*)d_in[2];
  const float* b1 = (const float*)d_in[3];
  const float* W2 = (const float*)d_in[4];
  const float* b2 = (const float*)d_in[5];

  const int HID = in_sizes[3];            // 256
  const int C   = in_sizes[5];            // 47
  const int FIN = in_sizes[2] / HID;      // 500
  const int N   = in_sizes[0] / FIN;      // 100000
  const int E   = in_sizes[1] / 2;        // 1600000
  const int CP  = 64;                     // padded class width

  const int* src = ei;
  const int* dst = ei + E;

  auto cdiv = [](long long a, long long b) { return (int)((a + b - 1) / b); };
  const int Mtiles = cdiv(N, TM);
  const int Mpad   = Mtiles * TM;                  // 100096
  const int Kpad1  = cdiv(FIN, TK) * TK;           // 512
  const int Kpad2  = HID;                          // 256 (already /32)

  // Workspace layout (float units); ~410 MB for reference shapes.
  float* ws = (float*)d_ws;
  size_t off = 0;
  float* deg  = ws + off; off += ((size_t)N + 255) & ~(size_t)255;
  float* h1   = ws + off; off += (size_t)N * HID;
  float* agg1 = ws + off; off += (size_t)N * HID;
  float* h2   = ws + off; off += (size_t)N * CP;
  float* agg2 = ws + off; off += (size_t)N * CP;
  __bf16* xb  = (__bf16*)(ws + off); off += (size_t)Mpad * Kpad1 / 2;
  __bf16* w1t = (__bf16*)(ws + off); off += (size_t)HID * Kpad1 / 2;
  __bf16* h1b = (__bf16*)(ws + off); off += (size_t)Mpad * HID / 2;
  __bf16* w2t = (__bf16*)(ws + off); off += (size_t)CP * Kpad2 / 2;
  (void)ws_size; (void)n_in; (void)out_size;

  const int B = 256;

  // Zero degree + atomic accumulators (must be redone every launch).
  const long long degPad = ((long long)N + 255) & ~255LL;
  gcn_zero4<<<cdiv(degPad / 4, B), B, 0, stream>>>(deg, degPad / 4);
  gcn_zero4<<<cdiv((long long)N * HID / 4, B), B, 0, stream>>>(agg1, (long long)N * HID / 4);
  gcn_zero4<<<cdiv((long long)N * CP / 4, B), B, 0, stream>>>(agg2, (long long)N * CP / 4);

  // Degrees -> dinv.
  gcn_deg<<<cdiv(E, B), B, 0, stream>>>(dst, deg, E);
  gcn_dinv<<<cdiv(N, B), B, 0, stream>>>(deg, N);

  // bf16 operand prep (zero K-pad; B matrices transposed to [n][k]).
  {
    const int sh = __builtin_ctz(Kpad1 / 4);       // 7
    const long long total = (long long)N << sh;
    gcn_cvt_x<<<cdiv(total, B), B, 0, stream>>>(x, xb, total, sh, FIN, FIN, Kpad1);
  }
  gcn_cvt_w_t<<<cdiv((long long)HID * Kpad1, B), B, 0, stream>>>(W1, w1t, FIN, HID, Kpad1, HID);
  gcn_cvt_w_t<<<cdiv((long long)CP * Kpad2, B), B, 0, stream>>>(W2, w2t, HID, C, Kpad2, CP);

  // Layer 1: h1 = x @ W1  (async-staged bf16 WMMA)
  {
    dim3 g(Mtiles, HID / TN);
    gcn_gemm_bf16_wmma<<<g, 256, 0, stream>>>(xb, Kpad1, w1t, Kpad1, h1, HID, N, Kpad1 / TK);
  }
  // agg1 += h1[src]*coef  (h1 = 102 MB, L2-resident gathers)
  {
    const int sh = __builtin_ctz(HID / 4);         // 6
    const long long total = (long long)E << sh;
    gcn_scatter<<<cdiv(total, B), B, 0, stream>>>(h1, src, dst, deg, agg1, total, sh, HID);
  }
  // h1 = relu(agg1 + h1*dinv^2 + b1); also emit bf16 copy for GEMM2.
  {
    const int sh = __builtin_ctz(HID / 4);
    const long long total = (long long)N << sh;
    gcn_fuse_relu<<<cdiv(total, B), B, 0, stream>>>(h1, h1b, agg1, deg, b1, total, sh, HID);
  }
  // Layer 2: h2 = h1 @ W2  (cols 47..63 zero-padded)
  {
    dim3 g(Mtiles, CP / TN);
    gcn_gemm_bf16_wmma<<<g, 256, 0, stream>>>(h1b, Kpad2, w2t, Kpad2, h2, CP, N, Kpad2 / TK);
  }
  // agg2 += h2[src]*coef
  {
    const int sh = __builtin_ctz(CP / 4);          // 4
    const long long total = (long long)E << sh;
    gcn_scatter<<<cdiv(total, B), B, 0, stream>>>(h2, src, dst, deg, agg2, total, sh, CP);
  }
  // Final: bias + self-loop + log_softmax -> d_out [N, C]
  gcn_logsoftmax<<<cdiv((long long)N * 32, B), B, 0, stream>>>(
      h2, agg2, deg, b2, (float*)d_out, N, C, CP);
}
